// AttentionGate_36404142801188
// MI455X (gfx1250) — matrix operand order repaired and checked by reading, not compiled, not run on hardware
//
#include <hip/hip_runtime.h>
#include <hip/hip_bf16.h>

typedef _Float16 half_t;
typedef __attribute__((ext_vector_type(16))) _Float16 v16h;
typedef __attribute__((ext_vector_type(8)))  float    v8f;

#define CC   384
#define NHD  12
#define HD   32
#define WSZ  7
#define NWIN 343      // 7^3 tokens per window
#define NP   352      // padded to 22*16
#define NWND 64       // (28/7)^3 windows
#define LTOK 21952    // 28^3

// ---- optional CDNA5 async global->LDS copies (guarded: build stays green) ----
#if defined(__has_builtin)
#  if __has_builtin(__builtin_amdgcn_global_load_async_to_lds_b128) && \
      __has_builtin(__builtin_amdgcn_s_wait_asynccnt)
#    define USE_ASYNC_LDS 1
#  endif
#endif
#ifndef USE_ASYNC_LDS
#  define USE_ASYNC_LDS 0
#endif

typedef int v4i_ __attribute__((vector_size(16)));
typedef __attribute__((address_space(1))) v4i_ gv4i;
typedef __attribute__((address_space(3))) v4i_ lv4i;

static __device__ inline void copy16_to_lds(const half_t* g, half_t* l) {
#if USE_ASYNC_LDS
  __builtin_amdgcn_global_load_async_to_lds_b128(
      (gv4i*)(uintptr_t)g, (lv4i*)l, 0, 0);
#else
  *(uint4*)l = *(const uint4*)g;
#endif
}
static __device__ inline void lds_copy_wait() {
#if USE_ASYNC_LDS
  __builtin_amdgcn_s_wait_asynccnt(0);
#endif
}

static __device__ inline v8f wmma_f16(v16h a, v16h b, v8f c) {
  return __builtin_amdgcn_wmma_f32_16x16x32_f16(false, a, false, b, (short)0, c,
                                                false, false);
}

union H2U  { unsigned u; half_t h[2]; };
union H2U4 { uint4 u4; half_t h[8]; };
union Frag { v16h v; unsigned u[8]; };

// A fragment: 16x32 (MxK), row-major source with stride ld (halfs).
static __device__ inline v16h load_a_frag(const half_t* base, int ld) {
  int lane = threadIdx.x & 31;
  int m = lane & 15;
  int hsel = lane >> 4;
  Frag f;
#pragma unroll
  for (int j = 0; j < 8; ++j) {
    int k = (j >> 2) * 16 + hsel * 8 + (j & 3) * 2;
    f.u[j] = *reinterpret_cast<const unsigned*>(base + m * ld + k);
  }
  return f.v;
}

// B fragment: 32x16 (KxN). Source stored N-major: base[n*ld + k] = B[k][n].
static __device__ inline v16h load_b_frag(const half_t* base, int ld) {
  int lane = threadIdx.x & 31;
  int n = lane & 15;
  int kg = lane >> 4;
  Frag f;
#pragma unroll
  for (int j = 0; j < 8; ++j) {
    int k = (j >> 2) * 16 + kg * 8 + (j & 3) * 2;
    f.u[j] = *reinterpret_cast<const unsigned*>(base + n * ld + k);
  }
  return f.v;
}

// ---------------------------------------------------------------- prep: LN1 + residual
__global__ __launch_bounds__(256) void prep_kernel(
    const float* __restrict__ skip, const float* __restrict__ xup,
    const float* __restrict__ g1, const float* __restrict__ b1,
    float* __restrict__ X, half_t* __restrict__ SKLN, half_t* __restrict__ QLN) {
  int l = blockIdx.x * 8 + (threadIdx.x >> 5);
  int lane = threadIdx.x & 31;
  const float* sp = skip + (size_t)l * CC;
  const float* xp = xup + (size_t)l * CC;
  float sv[12], xv[12];
  float ss = 0.f, ss2 = 0.f, sx = 0.f, sx2 = 0.f;
#pragma unroll
  for (int i = 0; i < 12; ++i) {
    int c = lane + i * 32;
    float a = sp[c], b = xp[c];
    sv[i] = a; xv[i] = b;
    ss += a; ss2 += a * a; sx += b; sx2 += b * b;
  }
#pragma unroll
  for (int off = 1; off < 32; off <<= 1) {
    ss += __shfl_xor(ss, off, 32);  ss2 += __shfl_xor(ss2, off, 32);
    sx += __shfl_xor(sx, off, 32);  sx2 += __shfl_xor(sx2, off, 32);
  }
  float ms = ss / CC, vs = ss2 / CC - ms * ms, rs = rsqrtf(vs + 1e-5f);
  float mx = sx / CC, vx = sx2 / CC - mx * mx, rx = rsqrtf(vx + 1e-5f);
  int d = l / 784, rem = l % 784, hh = rem / 28, ww = rem % 28;
  int wi = (d / WSZ) * 16 + (hh / WSZ) * 4 + (ww / WSZ);
  int n = (d % WSZ) * 49 + (hh % WSZ) * WSZ + (ww % WSZ);
  size_t dst = ((size_t)wi * NP + n) * CC;
  const float qscale = 0.17677669529663687f;  // hd^-0.5
#pragma unroll
  for (int i = 0; i < 12; ++i) {
    int c = lane + i * 32;
    float gg = g1[c], bb = b1[c];
    X[(size_t)l * CC + c] = sv[i] + xv[i];
    SKLN[dst + c] = (half_t)((sv[i] - ms) * rs * gg + bb);
    QLN[dst + c] = (half_t)(((xv[i] - mx) * rx * gg + bb) * qscale);
  }
}

__global__ __launch_bounds__(256) void padzero_kernel(half_t* SKLN, half_t* QLN) {
  int i = blockIdx.x * blockDim.x + threadIdx.x;
  const int total = NWND * (NP - NWIN) * CC;
  if (i < total) {
    int w = i / ((NP - NWIN) * CC);
    int r = (i / CC) % (NP - NWIN);
    int c = i % CC;
    size_t o = ((size_t)w * NP + NWIN + r) * CC + c;
    SKLN[o] = (half_t)0.f;
    QLN[o] = (half_t)0.f;
  }
}

// weight convert + transpose: dst[n*K + k] = (f16) src[k*N + n]
__global__ __launch_bounds__(256) void wcvtT_kernel(
    const float* __restrict__ src, half_t* __restrict__ dst, int K, int N) {
  int i = blockIdx.x * blockDim.x + threadIdx.x;
  if (i >= K * N) return;
  int n = i / K, k = i % K;
  dst[i] = (half_t)src[(size_t)k * N + n];
}

// bias table: BIASH[h][qn][kn] (f16), kn>=343 masked with -3e4
__global__ __launch_bounds__(256) void bias_kernel(
    const float* __restrict__ rpb, half_t* __restrict__ BIASH) {
  int i = blockIdx.x * blockDim.x + threadIdx.x;
  const int total = NHD * NP * NP;
  if (i >= total) return;
  int h = i / (NP * NP);
  int rem = i % (NP * NP);
  int qn = rem / NP, kn = rem % NP;
  float v;
  if (kn >= NWIN) {
    v = -30000.f;
  } else {
    int qc = qn < NWIN ? qn : NWIN - 1;
    int q0 = qc / 49, q1 = (qc / 7) % 7, q2 = qc % 7;
    int k0 = kn / 49, k1 = (kn / 7) % 7, k2 = kn % 7;
    int idx = ((q0 - k0 + 6) * 13 + (q1 - k1 + 6)) * 13 + (q2 - k2 + 6);
    v = rpb[idx * NHD + h];
  }
  BIASH[i] = (half_t)v;
}

__global__ __launch_bounds__(256) void ln2_kernel(
    const float* __restrict__ X, const float* __restrict__ g2,
    const float* __restrict__ b2, half_t* __restrict__ X2H) {
  int l = blockIdx.x * 8 + (threadIdx.x >> 5);
  int lane = threadIdx.x & 31;
  const float* xp = X + (size_t)l * CC;
  float v[12];
  float s = 0.f, s2 = 0.f;
#pragma unroll
  for (int i = 0; i < 12; ++i) {
    int c = lane + i * 32;
    float a = xp[c];
    v[i] = a; s += a; s2 += a * a;
  }
#pragma unroll
  for (int off = 1; off < 32; off <<= 1) {
    s += __shfl_xor(s, off, 32);  s2 += __shfl_xor(s2, off, 32);
  }
  float m = s / CC, var = s2 / CC - m * m, r = rsqrtf(var + 1e-5f);
#pragma unroll
  for (int i = 0; i < 12; ++i) {
    int c = lane + i * 32;
    X2H[(size_t)l * CC + c] = (half_t)((v[i] - m) * r * g2[c] + b2[c]);
  }
}

// ---------------------------------------------------------------- generic WMMA GEMM
// 64x128 tiles, 4 waves, each wave 32x64 (2x4 frags = 8 WMMA / K-step).
// Weights pre-transposed to N-major WT[n*K+k].
// MODE 0: KV scatter   MODE 1: proj + window-reverse residual into X
// MODE 2: GELU -> OH   MODE 3: residual X + val -> OF
template <int MODE>
__global__ __launch_bounds__(128) void gemm_kernel(
    const half_t* __restrict__ A, int lda, const half_t* __restrict__ WT,
    int K, const float* __restrict__ bias, half_t* KB, half_t* VB, float* X,
    half_t* OH, float* OF) {
  constexpr int BKP = 40;  // padded LDS stride for B
  __shared__ alignas(16) half_t Asm[64 * 32];
  __shared__ alignas(16) half_t Bsm[128 * BKP];
  int tid = threadIdx.x, wid = tid >> 5, lane = tid & 31;
  int n0 = blockIdx.x * 128, m0 = blockIdx.y * 64;
  int wm = wid >> 1, wn = wid & 1;
  v8f acc[2][4] = {};
  // per-thread staging coordinates
  int arow = tid >> 1, acg = tid & 1;            // A: 64 rows x 2 x 16B
  const half_t* asrc = A + (size_t)(m0 + arow) * lda + acg * 16;
  half_t* adst = Asm + arow * 32 + acg * 16;
  const half_t* bsrc = WT + (size_t)(n0 + tid) * K;  // B: 128 rows x 32 halfs
  half_t* bdst = Bsm + tid * BKP;
  for (int kk = 0; kk < K; kk += 32) {
    copy16_to_lds(asrc + kk, adst);
    copy16_to_lds(asrc + kk + 8, adst + 8);
    copy16_to_lds(bsrc + kk, bdst);
    copy16_to_lds(bsrc + kk + 8, bdst + 8);
    copy16_to_lds(bsrc + kk + 16, bdst + 16);
    copy16_to_lds(bsrc + kk + 24, bdst + 24);
    if (kk + 32 < K) {  // prefetch next K-step (global_prefetch_b8)
      __builtin_prefetch(asrc + kk + 32, 0, 3);
      __builtin_prefetch(bsrc + kk + 32, 0, 3);
    }
    lds_copy_wait();
    __syncthreads();
    v16h a0 = load_a_frag(Asm + (wm * 32) * 32, 32);
    v16h a1 = load_a_frag(Asm + (wm * 32 + 16) * 32, 32);
#pragma unroll
    for (int nt = 0; nt < 4; ++nt) {
      v16h b = load_b_frag(Bsm + (wn * 64 + nt * 16) * BKP, BKP);
      acc[0][nt] = wmma_f16(a0, b, acc[0][nt]);
      acc[1][nt] = wmma_f16(a1, b, acc[1][nt]);
    }
    __syncthreads();
  }
  int hi = lane >> 4, nl = lane & 15;
#pragma unroll
  for (int mt = 0; mt < 2; ++mt)
#pragma unroll
    for (int nt = 0; nt < 4; ++nt)
#pragma unroll
      for (int r = 0; r < 8; ++r) {
        int row = m0 + wm * 32 + mt * 16 + r + 8 * hi;
        int col = n0 + wn * 64 + nt * 16 + nl;
        float v = acc[mt][nt][r] + bias[col];
        if constexpr (MODE == 0) {
          int kv = col >= CC;
          int c = col - (kv ? CC : 0);
          int head = c >> 5, dd = c & 31;
          int w = row / NP, n = row % NP;
          half_t* dst = kv ? VB : KB;
          dst[(((size_t)(w * NHD + head)) * NP + n) * HD + dd] = (half_t)v;
        } else if constexpr (MODE == 1) {
          int w = row / NP, n = row % NP;
          if (n < NWIN) {
            int wd = w >> 4, wh = (w >> 2) & 3, wwn = w & 3;
            int id = n / 49, ih = (n / 7) % 7, iw = n % 7;
            int l = ((wd * 7 + id) * 28 + (wh * 7 + ih)) * 28 + (wwn * 7 + iw);
            size_t o = (size_t)l * CC + col;
            X[o] += v;
          }
        } else if constexpr (MODE == 2) {
          float t = v;
          float g = 0.5f * t *
                    (1.0f + tanhf(0.7978845608f * (t + 0.044715f * t * t * t)));
          OH[(size_t)row * 1536 + col] = (half_t)g;
        } else {
          size_t o = (size_t)row * CC + col;
          OF[o] = X[o] + v;
        }
      }
}

// ---------------------------------------------------------------- flash attention
// Transposed formulation: S^T = K @ Q^T, o^T = V^T @ P^T.
__global__ __launch_bounds__(128) void attn_kernel(
    const half_t* __restrict__ QLN, const half_t* __restrict__ KB,
    const half_t* __restrict__ VB, const half_t* __restrict__ BIASH,
    half_t* __restrict__ AO) {
  int blk = blockIdx.x;
  int w = blk / NHD, h = blk % NHD;
  int tid = threadIdx.x, wid = tid >> 5, lane = tid & 31;
  __shared__ alignas(16) half_t Ksm[NP * HD];     // K natural [kn][d]
  __shared__ alignas(16) half_t Vt[HD * NP];      // V transposed [d][kn]
  __shared__ alignas(16) half_t Psm[4][16 * 32];  // per-wave P [qn][kn]
  const half_t* Kg = KB + (size_t)(w * NHD + h) * NP * HD;
  const half_t* Vg = VB + (size_t)(w * NHD + h) * NP * HD;
  {
    for (int i = tid; i < NP * HD / 8; i += 128)
      copy16_to_lds(Kg + i * 8, Ksm + i * 8);
    const unsigned* Vg2 = (const unsigned*)Vg;
    for (int i = tid; i < NP * HD / 2; i += 128) {
      int kn = i >> 4, d = (i & 15) * 2;
      H2U cv; cv.u = Vg2[i];
      Vt[d * NP + kn] = cv.h[0];
      Vt[(d + 1) * NP + kn] = cv.h[1];
    }
    lds_copy_wait();
  }
  __syncthreads();
  int hi = lane >> 4, nl = lane & 15;
  half_t* Pw = Psm[wid];
  for (int qt = wid; qt < NP / 16; qt += 4) {
    int qn = qt * 16 + nl;  // this lane's query column
    v16h qb = load_b_frag(QLN + ((size_t)(w * NP + qt * 16)) * CC + h * HD, CC);
    const half_t* brow = BIASH + ((size_t)h * NP + qn) * NP;
    v8f acc0 = {}, acc1 = {};  // o^T: rows d (acc0: 0-15, acc1: 16-31), cols qn
    float rm = -1e30f, rsum = 0.f;
    for (int kt = 0; kt < NP / 32; ++kt) {
      int kn0 = kt * 32;
      v16h ka0 = load_a_frag(Ksm + kn0 * HD, HD);
      v16h ka1 = load_a_frag(Ksm + (kn0 + 16) * HD, HD);
      v8f z = {};
      v8f s0 = wmma_f16(ka0, qb, z);  // S^T rows kn0+hi*8+r
      v8f s1 = wmma_f16(ka1, qb, z);  // S^T rows kn0+16+hi*8+r
      H2U4 b0, b1;
      b0.u4 = *(const uint4*)(brow + kn0 + hi * 8);
      b1.u4 = *(const uint4*)(brow + kn0 + 16 + hi * 8);
      float v0[8], v1[8];
      float mx = -1e30f;
#pragma unroll
      for (int r = 0; r < 8; ++r) {
        v0[r] = s0[r] + (float)b0.h[r];
        v1[r] = s1[r] + (float)b1.h[r];
        mx = fmaxf(mx, fmaxf(v0[r], v1[r]));
      }
      mx = fmaxf(mx, __shfl_xor(mx, 16, 32));  // combine kn halves
      float nm = fmaxf(rm, mx);
      float sc = __expf(rm - nm);
      rm = nm;
      H2U4 P0, P1;
      float ls = 0.f;
#pragma unroll
      for (int r = 0; r < 8; ++r) {
        float p0 = __expf(v0[r] - nm);
        float p1 = __expf(v1[r] - nm);
        P0.h[r] = (half_t)p0;
        P1.h[r] = (half_t)p1;
        ls += p0 + p1;
      }
      ls += __shfl_xor(ls, 16, 32);
      rsum = rsum * sc + ls;
#pragma unroll
      for (int r = 0; r < 8; ++r) { acc0[r] *= sc; acc1[r] *= sc; }
      *(uint4*)(Pw + nl * 32 + hi * 8) = P0.u4;
      *(uint4*)(Pw + nl * 32 + 16 + hi * 8) = P1.u4;
      v16h pb = load_b_frag(Pw, 32);                    // P^T as B
      v16h va0 = load_a_frag(Vt + kn0, NP);             // V^T rows d 0-15
      v16h va1 = load_a_frag(Vt + 16 * NP + kn0, NP);   // V^T rows d 16-31
      acc0 = wmma_f16(va0, pb, acc0);
      acc1 = wmma_f16(va1, pb, acc1);
    }
    float inv = 1.0f / rsum;
    size_t base = ((size_t)(w * NP + qn)) * CC + h * HD;
    H2U4 o0, o1;
#pragma unroll
    for (int r = 0; r < 8; ++r) {
      o0.h[r] = (half_t)(acc0[r] * inv);  // d = hi*8 + r
      o1.h[r] = (half_t)(acc1[r] * inv);  // d = 16 + hi*8 + r
    }
    *(uint4*)(AO + base + hi * 8) = o0.u4;
    *(uint4*)(AO + base + 16 + hi * 8) = o1.u4;
  }
}

// ---------------------------------------------------------------- host launcher
extern "C" void kernel_launch(void* const* d_in, const int* in_sizes, int n_in,
                              void* d_out, int out_size, void* d_ws,
                              size_t ws_size, hipStream_t stream) {
  const float* skip   = (const float*)d_in[0];
  const float* x_up   = (const float*)d_in[1];
  const float* ln1_g  = (const float*)d_in[5];
  const float* ln1_b  = (const float*)d_in[6];
  const float* kv_w   = (const float*)d_in[7];
  const float* kv_b   = (const float*)d_in[8];
  const float* rpb    = (const float*)d_in[9];
  const float* proj_w = (const float*)d_in[10];
  const float* proj_b = (const float*)d_in[11];
  const float* ln2_g  = (const float*)d_in[12];
  const float* ln2_b  = (const float*)d_in[13];
  const float* mlp_w1 = (const float*)d_in[14];
  const float* mlp_b1 = (const float*)d_in[15];
  const float* mlp_w2 = (const float*)d_in[16];
  const float* mlp_b2 = (const float*)d_in[17];

  char* ws = (char*)d_ws;
  size_t off = 0;
  auto alloc = [&](size_t bytes) {
    size_t o = off;
    off = (off + bytes + 255) & ~(size_t)255;
    return o;
  };
  float*  X     = (float*)(ws + alloc((size_t)LTOK * CC * 4));
  half_t* SKLN  = (half_t*)(ws + alloc((size_t)NWND * NP * CC * 2));
  half_t* QLN   = (half_t*)(ws + alloc((size_t)NWND * NP * CC * 2));
  half_t* KB    = (half_t*)(ws + alloc((size_t)NWND * NHD * NP * HD * 2));
  half_t* VB    = (half_t*)(ws + alloc((size_t)NWND * NHD * NP * HD * 2));
  half_t* AO    = (half_t*)(ws + alloc((size_t)NWND * NP * CC * 2));
  half_t* X2H   = (half_t*)(ws + alloc((size_t)LTOK * CC * 2));
  half_t* H1    = (half_t*)(ws + alloc((size_t)LTOK * 1536 * 2));
  half_t* BIASH = (half_t*)(ws + alloc((size_t)NHD * NP * NP * 2));
  half_t* Wkv   = (half_t*)(ws + alloc((size_t)294912 * 2));
  half_t* Wpr   = (half_t*)(ws + alloc((size_t)147456 * 2));
  half_t* Wm1   = (half_t*)(ws + alloc((size_t)589824 * 2));
  half_t* Wm2   = (half_t*)(ws + alloc((size_t)589824 * 2));

  // weights: convert + transpose to N-major [n*K+k]
  wcvtT_kernel<<<1152, 256, 0, stream>>>(kv_w,   Wkv, 384, 768);
  wcvtT_kernel<<<576,  256, 0, stream>>>(proj_w, Wpr, 384, 384);
  wcvtT_kernel<<<2304, 256, 0, stream>>>(mlp_w1, Wm1, 384, 1536);
  wcvtT_kernel<<<2304, 256, 0, stream>>>(mlp_w2, Wm2, 1536, 384);
  bias_kernel<<<(NHD * NP * NP + 255) / 256, 256, 0, stream>>>(rpb, BIASH);
  prep_kernel<<<LTOK / 8, 256, 0, stream>>>(skip, x_up, ln1_g, ln1_b, X, SKLN, QLN);
  padzero_kernel<<<864, 256, 0, stream>>>(SKLN, QLN);
  // kv = skln @ kv_w + kv_b  (M=22528, K=384, N=768) -> KB/VB
  gemm_kernel<0><<<dim3(768 / 128, (NWND * NP) / 64), 128, 0, stream>>>(
      SKLN, CC, Wkv, CC, kv_b, KB, VB, nullptr, nullptr, nullptr);
  attn_kernel<<<NWND * NHD, 128, 0, stream>>>(QLN, KB, VB, BIASH, AO);
  // proj + residual into X (M=22528, K=384, N=384)
  gemm_kernel<1><<<dim3(CC / 128, (NWND * NP) / 64), 128, 0, stream>>>(
      AO, CC, Wpr, CC, proj_b, nullptr, nullptr, X, nullptr, nullptr);
  ln2_kernel<<<LTOK / 8, 256, 0, stream>>>(X, ln2_g, ln2_b, X2H);
  // h1 = gelu(x2h @ w1 + b1) (M=21952, K=384, N=1536)
  gemm_kernel<2><<<dim3(1536 / 128, LTOK / 64), 128, 0, stream>>>(
      X2H, CC, Wm1, CC, mlp_b1, nullptr, nullptr, nullptr, H1, nullptr);
  // out = X + h1 @ w2 + b2 (M=21952, K=1536, N=384)
  gemm_kernel<3><<<dim3(CC / 128, LTOK / 64), 128, 0, stream>>>(
      H1, 1536, Wm2, 1536, mlp_b2, nullptr, nullptr, X, nullptr,
      (float*)d_out);
}
